// YugiwarabeNet_58815282152341
// MI455X (gfx1250) — compile-verified, matrix operand order checked by reading, not run on hardware
//
#include <hip/hip_runtime.h>
#include <hip/hip_bf16.h>

typedef __attribute__((ext_vector_type(16))) __bf16       v16bf;
typedef __attribute__((ext_vector_type(8)))  float        v8f;
typedef __attribute__((ext_vector_type(8)))  unsigned int v8u;

#define R_ROWS 41472   // 512 boards * 81 tokens
#define NBOARD 512

__device__ __forceinline__ unsigned short f2bf(float f) {
    unsigned int u = __builtin_bit_cast(unsigned int, f);
    u += 0x7FFFu + ((u >> 16) & 1u);   // round-to-nearest-even
    return (unsigned short)(u >> 16);
}

// ---------------------------------------------------------------------------
// fp32 -> bf16 conversion kernels (flat, and transpose K x N -> N x K)
// ---------------------------------------------------------------------------
__global__ void cvt_kernel(const float* __restrict__ in, unsigned short* __restrict__ out, int n)
{
    int idx = blockIdx.x * 256 + threadIdx.x;
    if (idx < n) out[idx] = f2bf(in[idx]);
}

__global__ void cvtT_kernel(const float* __restrict__ in, unsigned short* __restrict__ out,
                            int K, int N)   // in: K x N, out: N x K
{
    int idx = blockIdx.x * 256 + threadIdx.x;
    if (idx >= K * N) return;
    int nn = idx / K, k = idx % K;
    out[idx] = f2bf(in[(size_t)k * N + nn]);
}

// ---------------------------------------------------------------------------
// bf16 WMMA GEMM:  C[M,N] = act( (A[M,K(lda)] * B[N,K]) * scale + bias + res )
// A, B bf16 row-major with K contiguous. block = 128 threads (4 waves),
// tile 64(M) x 128(N), K-step 32, LDS double-buffered via async-to-LDS DMA.
// Each wave computes a 32x64 sub-tile = 2x4 WMMA 16x16 tiles.
// ---------------------------------------------------------------------------
__global__ __launch_bounds__(128)
void wmma_gemm_kernel(const unsigned short* __restrict__ A, const unsigned short* __restrict__ B,
                      const float* __restrict__ bias, const float* __restrict__ scale,
                      const float* __restrict__ res, float* __restrict__ C,
                      int M, int N, int K, int lda, int act)
{
    __shared__ __align__(16) unsigned short As[2][64 * 40];
    __shared__ __align__(16) unsigned short Bs[2][128 * 40];

    const int t    = threadIdx.x;
    const int lane = t & 31;
    const int wave = t >> 5;
    const int m0   = blockIdx.y * 64;
    const int n0   = blockIdx.x * 128;
    const int wm   = (wave >> 1) * 32;
    const int wn   = (wave & 1) * 64;
    const int half = lane >> 4;
    const int l16  = lane & 15;

    v8f acc[2][4] = {};

    // fragment-load + 8x WMMA from LDS buffer `buf`
    auto compute_step = [&](int buf) {
        v8u au[2], bu[4];
        #pragma unroll
        for (int mi = 0; mi < 2; mi++) {
            int mrow = wm + mi * 16 + l16;
            #pragma unroll
            for (int v = 0; v < 8; v++) {
                int kb = ((v < 4) ? (2 * v) : (2 * (v - 4) + 16)) + 8 * half;
                au[mi][v] = *(const unsigned int*)&As[buf][mrow * 40 + kb];
            }
        }
        #pragma unroll
        for (int ni = 0; ni < 4; ni++) {
            int nrow = wn + ni * 16 + l16;
            #pragma unroll
            for (int v = 0; v < 8; v++) {
                int kb = ((v < 4) ? (2 * v) : (2 * (v - 4) + 16)) + 8 * half;
                bu[ni][v] = *(const unsigned int*)&Bs[buf][nrow * 40 + kb];
            }
        }
        #pragma unroll
        for (int mi = 0; mi < 2; mi++)
            #pragma unroll
            for (int ni = 0; ni < 4; ni++)
                acc[mi][ni] = __builtin_amdgcn_wmma_f32_16x16x32_bf16(
                    false, __builtin_bit_cast(v16bf, au[mi]),
                    false, __builtin_bit_cast(v16bf, bu[ni]),
                    (short)0, acc[mi][ni], false, false);
    };

    const bool fast = (m0 + 64 <= M) && (n0 + 128 <= N) && ((K & 31) == 0);

    if (fast) {
        const unsigned asb = (unsigned)(size_t)&As[0][0];
        const unsigned bsb = (unsigned)(size_t)&Bs[0][0];
        const unsigned abufstride = 64u * 40u * 2u;
        const unsigned bbufstride = 128u * 40u * 2u;

        // async-DMA a (rows x 32) bf16 tile into LDS, 16B chunks, 4 chunks/row
        auto stage = [&](const unsigned short* base, int row0, int ld, int k0,
                         unsigned ldsbase, int reps) {
            for (int u = 0; u < reps; u++) {
                int chunk = u * 128 + t;
                int row = chunk >> 2, q = chunk & 3;
                unsigned lo = ldsbase + (unsigned)(row * 80 + q * 16);
                unsigned long long ga = (unsigned long long)(const void*)
                    (base + (size_t)(row0 + row) * ld + k0 + q * 8);
                asm volatile("global_load_async_to_lds_b128 %0, %1, off"
                             :: "v"(lo), "v"(ga) : "memory");
            }
        };

        stage(A, m0, lda, 0, asb, 2);
        stage(B, n0, K,   0, bsb, 4);
        asm volatile("s_wait_asynccnt 0x0" ::: "memory");
        __syncthreads();
        for (int k0 = 0; k0 < K; k0 += 32) {
            int cur = (k0 >> 5) & 1;
            bool more = (k0 + 32) < K;
            if (more) {                                 // DMA next tile over WMMAs
                stage(A, m0, lda, k0 + 32, asb + (unsigned)(cur ^ 1) * abufstride, 2);
                stage(B, n0, K,   k0 + 32, bsb + (unsigned)(cur ^ 1) * bbufstride, 4);
            }
            compute_step(cur);
            asm volatile("s_wait_asynccnt 0x0" ::: "memory");
            __syncthreads();
        }
    } else {
        for (int k0 = 0; k0 < K; k0 += 32) {
            for (int e = t; e < 64 * 32; e += 128) {
                int m = e >> 5, k = e & 31;
                int gm = m0 + m, gk = k0 + k;
                As[0][m * 40 + k] = (gm < M && gk < K) ? A[(size_t)gm * lda + gk]
                                                       : (unsigned short)0;
            }
            for (int e = t; e < 128 * 32; e += 128) {
                int n = e >> 5, k = e & 31;
                int gn = n0 + n, gk = k0 + k;
                Bs[0][n * 40 + k] = (gn < N && gk < K) ? B[(size_t)gn * K + gk]
                                                       : (unsigned short)0;
            }
            __syncthreads();
            compute_step(0);
            __syncthreads();
        }
    }

    #pragma unroll
    for (int mi = 0; mi < 2; mi++)
        #pragma unroll
        for (int ni = 0; ni < 4; ni++) {
            int nn = n0 + wn + ni * 16 + l16;
            if (nn >= N) continue;
            float sc = scale ? scale[nn] : 1.0f;
            float bi = bias ? bias[nn] : 0.0f;
            #pragma unroll
            for (int r = 0; r < 8; r++) {
                int mm = m0 + wm + mi * 16 + r + 8 * half;
                if (mm >= M) continue;
                float v = acc[mi][ni][r] * sc + bi;
                if (res) v += res[(size_t)mm * N + nn];
                if (act) v = fmaxf(v, 0.0f);
                C[(size_t)mm * N + nn] = v;
            }
        }
}

// ---------------------------------------------------------------------------
// generic weight transpose: in[O][R] -> out[R][O]  (fp32, for direct convs)
// ---------------------------------------------------------------------------
__global__ void transpose_kernel(const float* __restrict__ in, float* __restrict__ out,
                                 int O, int R)
{
    int idx = blockIdx.x * 256 + threadIdx.x;
    if (idx >= O * R) return;
    int o = idx / R, r = idx % R;
    out[r * O + o] = in[o * R + r];
}

// ---------------------------------------------------------------------------
// stem: 3x3 conv 104->256 over 9x9, BN + relu, out layout (n*81+p, 256)
// ---------------------------------------------------------------------------
__global__ __launch_bounds__(256)
void stem_conv_kernel(const float* __restrict__ x, const float* __restrict__ wT,
                      const float* __restrict__ s, const float* __restrict__ b,
                      float* __restrict__ out)
{
    __shared__ float patch[936];
    int row = blockIdx.x;
    int n = row / 81, p = row % 81;
    int y = p / 9, x0 = p % 9;
    int t = threadIdx.x;
    for (int e = t; e < 936; e += 256) {
        int c = e / 9, tap = e % 9;
        int ky = tap / 3, kx = tap % 3;
        int yy = y + ky - 1, xc = x0 + kx - 1;
        float v = 0.0f;
        if (yy >= 0 && yy < 9 && xc >= 0 && xc < 9) v = x[(n * 104 + c) * 81 + yy * 9 + xc];
        patch[e] = v;
    }
    __syncthreads();
    float acc = 0.0f;
    for (int r = 0; r < 936; r++) acc += patch[r] * wT[r * 256 + t];
    out[row * 256 + t] = fmaxf(acc * s[t] + b[t], 0.0f);
}

// ---------------------------------------------------------------------------
// grouped 3x3 conv (32 groups of 4 ch, 128 ch), BN + relu, (row,128) layout
// ---------------------------------------------------------------------------
__global__ __launch_bounds__(128)
void gconv_kernel(const float* __restrict__ in, const float* __restrict__ wT,
                  const float* __restrict__ s, const float* __restrict__ b,
                  float* __restrict__ out)
{
    __shared__ float patch[9 * 128];
    int row = blockIdx.x;
    int n = row / 81, p = row % 81;
    int y = p / 9, x0 = p % 9;
    int t = threadIdx.x;
    for (int e = t; e < 1152; e += 128) {
        int tap = e >> 7, c = e & 127;
        int ky = tap / 3, kx = tap % 3;
        int yy = y + ky - 1, xc = x0 + kx - 1;
        float v = 0.0f;
        if (yy >= 0 && yy < 9 && xc >= 0 && xc < 9) v = in[(n * 81 + yy * 9 + xc) * 128 + c];
        patch[tap * 128 + c] = v;
    }
    __syncthreads();
    int g = t >> 2;
    float acc = 0.0f;
    #pragma unroll
    for (int ic = 0; ic < 4; ic++)
        #pragma unroll
        for (int tap = 0; tap < 9; tap++)
            acc += patch[tap * 128 + g * 4 + ic] * wT[(ic * 9 + tap) * 128 + t];
    out[row * 128 + t] = fmaxf(acc * s[t] + b[t], 0.0f);
}

__global__ void posadd_kernel(float* __restrict__ F, const float* __restrict__ pos)
{
    int row = blockIdx.x, c = threadIdx.x;
    F[row * 256 + c] += pos[(row % 81) * 256 + c];
}

__global__ void mean81_kernel(const float* __restrict__ in, float* __restrict__ out)
{
    int n = blockIdx.x, c = threadIdx.x;
    float s = 0.0f;
    for (int p = 0; p < 81; p++) s += in[(n * 81 + p) * 256 + c];
    out[n * 256 + c] = s * (1.0f / 81.0f);
}

// GNN fused: F = relu(LN(F + t1 + t2[row/81]) * s + b)   (in place)
__global__ __launch_bounds__(256)
void gnn_fuse_kernel(float* __restrict__ F, const float* __restrict__ t1,
                     const float* __restrict__ t2,
                     const float* __restrict__ s, const float* __restrict__ b)
{
    __shared__ float red[256];
    __shared__ float mv[2];
    int row = blockIdx.x, c = threadIdx.x;
    float v = F[row * 256 + c] + t1[row * 256 + c] + t2[(row / 81) * 256 + c];
    red[c] = v; __syncthreads();
    for (int o = 128; o > 0; o >>= 1) { if (c < o) red[c] += red[c + o]; __syncthreads(); }
    if (c == 0) mv[0] = red[0] * (1.0f / 256.0f);
    __syncthreads();
    float d = v - mv[0];
    red[c] = d * d; __syncthreads();
    for (int o = 128; o > 0; o >>= 1) { if (c < o) red[c] += red[c + o]; __syncthreads(); }
    if (c == 0) mv[1] = red[0] * (1.0f / 256.0f);
    __syncthreads();
    F[row * 256 + c] = fmaxf(d * rsqrtf(mv[1] + 1e-5f) * s[c] + b[c], 0.0f);
}

__global__ __launch_bounds__(256)
void ln_kernel(const float* __restrict__ in, const float* __restrict__ s,
               const float* __restrict__ b, float* __restrict__ out)
{
    __shared__ float red[256];
    __shared__ float mv[2];
    int row = blockIdx.x, c = threadIdx.x;
    float v = in[row * 256 + c];
    red[c] = v; __syncthreads();
    for (int o = 128; o > 0; o >>= 1) { if (c < o) red[c] += red[c + o]; __syncthreads(); }
    if (c == 0) mv[0] = red[0] * (1.0f / 256.0f);
    __syncthreads();
    float d = v - mv[0];
    red[c] = d * d; __syncthreads();
    for (int o = 128; o > 0; o >>= 1) { if (c < o) red[c] += red[c + o]; __syncthreads(); }
    if (c == 0) mv[1] = red[0] * (1.0f / 256.0f);
    __syncthreads();
    out[row * 256 + c] = d * rsqrtf(mv[1] + 1e-5f) * s[c] + b[c];
}

// ---------------------------------------------------------------------------
// MHA core: one block per (batch, head). qkv: (Nb*L, 768), out: (Nb*L, 256)
// biasmode: 0=none, 1=ray mask (computed), 2=piece mask pm[(b*81+i)*81+j]
// ---------------------------------------------------------------------------
__device__ __forceinline__ float attn_bias(int mode, const float* __restrict__ pm,
                                           int b, int i, int j)
{
    if (mode == 1) {
        int ri = i / 9, ci = i % 9, rj = j / 9, cj = j % 9;
        int dr = ri - rj; if (dr < 0) dr = -dr;
        int dc = ci - cj; if (dc < 0) dc = -dc;
        bool same = (ri == rj) || (ci == cj) || (dr == dc);
        return same ? 0.0f : -1e9f;
    } else if (mode == 2) {
        return pm[(b * 81 + i) * 81 + j];
    }
    return 0.0f;
}

__global__ __launch_bounds__(128)
void attention_kernel(const float* __restrict__ qkv, const float* __restrict__ pm,
                      float* __restrict__ out, int L, int biasmode)
{
    __shared__ float Qs[81 * 33];
    __shared__ float Ks[81 * 33];
    __shared__ float Vs[81 * 33];
    int b = blockIdx.x >> 3;
    int h = blockIdx.x & 7;
    int t = threadIdx.x;
    for (int e = t; e < L * 32; e += 128) {
        int r = e >> 5, d = e & 31;
        int base = (b * L + r) * 768 + h * 32 + d;
        Qs[r * 33 + d] = qkv[base];
        Ks[r * 33 + d] = qkv[base + 256];
        Vs[r * 33 + d] = qkv[base + 512];
    }
    __syncthreads();
    int i = t;
    if (i < L) {
        const float scl = 0.17677669529663687f;  // 1/sqrt(32)
        float mx = -1e30f, sum = 0.0f;
        for (int j = 0; j < L; j++) {
            float sc = 0.0f;
            #pragma unroll
            for (int d = 0; d < 32; d++) sc += Qs[i * 33 + d] * Ks[j * 33 + d];
            sc = sc * scl + attn_bias(biasmode, pm, b, i, j);
            float nm = fmaxf(mx, sc);
            sum = sum * __expf(mx - nm) + __expf(sc - nm);
            mx = nm;
        }
        float o[32];
        #pragma unroll
        for (int d = 0; d < 32; d++) o[d] = 0.0f;
        for (int j = 0; j < L; j++) {
            float sc = 0.0f;
            #pragma unroll
            for (int d = 0; d < 32; d++) sc += Qs[i * 33 + d] * Ks[j * 33 + d];
            sc = sc * scl + attn_bias(biasmode, pm, b, i, j);
            float p = __expf(sc - mx) / sum;
            #pragma unroll
            for (int d = 0; d < 32; d++) o[d] += p * Vs[j * 33 + d];
        }
        #pragma unroll
        for (int d = 0; d < 32; d++) out[(b * L + i) * 256 + h * 32 + d] = o[d];
    }
}

__global__ __launch_bounds__(256)
void logsoftmax_kernel(float* __restrict__ x, int N)
{
    __shared__ float red[256];
    __shared__ float mv[2];
    int row = blockIdx.x, t = threadIdx.x;
    float mx = -1e30f;
    for (int j = t; j < N; j += 256) mx = fmaxf(mx, x[row * N + j]);
    red[t] = mx; __syncthreads();
    for (int o = 128; o > 0; o >>= 1) { if (t < o) red[t] = fmaxf(red[t], red[t + o]); __syncthreads(); }
    if (t == 0) mv[0] = red[0];
    __syncthreads();
    mx = mv[0];
    float s = 0.0f;
    for (int j = t; j < N; j += 256) s += __expf(x[row * N + j] - mx);
    red[t] = s; __syncthreads();
    for (int o = 128; o > 0; o >>= 1) { if (t < o) red[t] += red[t + o]; __syncthreads(); }
    if (t == 0) mv[1] = mx + logf(red[0]);
    __syncthreads();
    float ls = mv[1];
    for (int j = t; j < N; j += 256) x[row * N + j] -= ls;
}

__global__ __launch_bounds__(64)
void value_kernel(const float* __restrict__ y, const float* __restrict__ w1,
                  const float* __restrict__ b1, const float* __restrict__ w2,
                  const float* __restrict__ b2, float* __restrict__ out)
{
    __shared__ float red[64];
    int bb = blockIdx.x, j = threadIdx.x;
    const float* last = y + (bb * 8 + 7) * 256;
    float a = 0.0f;
    for (int d = 0; d < 256; d++) a += last[d] * w1[d * 64 + j];
    a = fmaxf(a + b1[j], 0.0f);
    red[j] = a * w2[j];
    __syncthreads();
    for (int o = 32; o > 0; o >>= 1) { if (j < o) red[j] += red[j + o]; __syncthreads(); }
    if (j == 0) out[bb] = tanhf(red[0] + b2[0]);
}

__global__ __launch_bounds__(192)
void aux_kernel(const float* __restrict__ y, const float* __restrict__ w,
                const float* __restrict__ b, float* __restrict__ out)
{
    int bb = blockIdx.x;
    int k = threadIdx.x >> 5, lane = threadIdx.x & 31;
    if (k < 6) {
        const float* last = y + (bb * 8 + 7) * 256;
        float s = 0.0f;
        for (int d = lane; d < 256; d += 32) s += last[d] * w[k * 256 + d];
        for (int o = 16; o > 0; o >>= 1) s += __shfl_xor(s, o, 32);
        if (lane == 0) out[bb * 6 + k] = s + b[k];
    }
}

// ---------------------------------------------------------------------------
extern "C" void kernel_launch(void* const* d_in, const int* in_sizes, int n_in,
                              void* d_out, int out_size, void* d_ws, size_t ws_size,
                              hipStream_t stream)
{
    (void)in_sizes; (void)n_in; (void)out_size; (void)ws_size;
    const float* x_seq = (const float*)d_in[0];
    const float* pmask = (const float*)d_in[1];
#define PP(i) ((const float*)d_in[2 + (i)])
    const float* stem_w = PP(0);  const float* stem_s = PP(1);  const float* stem_b = PP(2);
    const float* c1_w = PP(3);    const float* bn1_s = PP(4);   const float* bn1_b = PP(5);
    const float* c2_w = PP(6);    const float* bn2_s = PP(7);   const float* bn2_b = PP(8);
    const float* c3_w = PP(9);    const float* bn3_s = PP(10);  const float* bn3_b = PP(11);
    const float* pos = PP(12);
    const float* gmsg_w = PP(13); const float* gmsg_b = PP(14);
    const float* gupd_w = PP(15); const float* gupd_b = PP(16);
    const float* gln_s = PP(17);  const float* gln_b = PP(18);
    const float* bqkv_w = PP(19); const float* bqkv_b = PP(20);
    const float* bout_w = PP(21); const float* bout_b = PP(22);
    const float* bln_s = PP(23);  const float* bln_b = PP(24);
    const float* bf1_w = PP(25);  const float* bf1_b = PP(26);
    const float* bf2_w = PP(27);  const float* bf2_b = PP(28);
    const float* tin_w = PP(29);  const float* tin_b = PP(30);
    const float* tout_w = PP(31); const float* tout_b = PP(32);
    const float* tln1_s = PP(33); const float* tln1_b = PP(34);
    const float* tln2_s = PP(35); const float* tln2_b = PP(36);
    const float* tf1_w = PP(37);  const float* tf1_b = PP(38);
    const float* tf2_w = PP(39);  const float* tf2_b = PP(40);
    const float* pol_w = PP(41);  const float* pol_b = PP(42);
    const float* vw1 = PP(43);    const float* vb1 = PP(44);
    const float* vw2 = PP(45);    const float* vb2 = PP(46);
    const float* aux_w = PP(47);  const float* aux_b = PP(48);
#undef PP

    const int R = R_ROWS;
    float* ws = (float*)d_ws;
    size_t off = 0;
    float* F    = ws + off; off += (size_t)R * 256;
    float* XN   = ws + off; off += (size_t)R * 256;
    float* QKV  = ws + off; off += (size_t)R * 768;
    float* T1   = ws + off; off += (size_t)R * 1024;
    float* MSG1 = ws + off; off += (size_t)NBOARD * 256;
    float* MSG2 = ws + off; off += (size_t)NBOARD * 256;
    float* Y    = ws + off; off += (size_t)NBOARD * 256;
    float* YT   = ws + off; off += (size_t)NBOARD * 2048;
    float* YQ   = ws + off; off += (size_t)NBOARD * 768;
    float* YA   = ws + off; off += (size_t)NBOARD * 256;
    float* WT   = ws + off; off += (size_t)936 * 256;
    float* WT2  = ws + off; off += (size_t)4 * 36 * 128 + 64;
    float* OB   = T1;   // attention output aliases FFN scratch
    float* T2   = QKV;  // grouped-conv output aliases qkv scratch

    // bf16 arenas (all B matrices stored N x K row-major, K contiguous)
    unsigned short* wb = (unsigned short*)(ws + off);
    size_t wo = 0;
    auto abf = [&](size_t n) { unsigned short* p = wb + wo; wo += n; return p; };
    unsigned short* c1bf   = abf((size_t)4 * 128 * 256);
    unsigned short* c3bf   = abf((size_t)4 * 256 * 128);
    unsigned short* gmsgbf = abf((size_t)4 * 256 * 256);
    unsigned short* gtopbf = abf((size_t)4 * 256 * 256);
    unsigned short* gbotbf = abf((size_t)4 * 256 * 256);
    unsigned short* qkvbf  = abf((size_t)24 * 768 * 256);
    unsigned short* outbf  = abf((size_t)24 * 256 * 256);
    unsigned short* f1bf   = abf((size_t)8 * 1024 * 256);
    unsigned short* f2wbf  = abf((size_t)8 * 256 * 1024);
    unsigned short* tinbf  = abf((size_t)4 * 768 * 256);
    unsigned short* toutbf = abf((size_t)4 * 256 * 256);
    unsigned short* tf1bf  = abf((size_t)4 * 2048 * 256);
    unsigned short* tf2bf  = abf((size_t)4 * 256 * 2048);
    unsigned short* polbf  = abf((size_t)2187 * 256 + 128);
    unsigned short* ABF    = abf((size_t)R * 1024);     // A-matrix bf16 scratch

    auto cvt = [&](const float* in, unsigned short* out, int n) {
        cvt_kernel<<<(n + 255) / 256, 256, 0, stream>>>(in, out, n);
    };
    auto cvtT = [&](const float* in, unsigned short* out, int K, int N) {
        cvtT_kernel<<<(K * N + 255) / 256, 256, 0, stream>>>(in, out, K, N);
    };
    auto gemm = [&](const unsigned short* A, const unsigned short* B, const float* bias,
                    const float* scale, const float* res, float* C,
                    int M, int N, int K, int lda, int act) {
        dim3 g((N + 127) / 128, (M + 63) / 64);
        wmma_gemm_kernel<<<g, dim3(128), 0, stream>>>(A, B, bias, scale, res, C,
                                                      M, N, K, lda, act);
    };
    auto gemmA = [&](const float* A, const unsigned short* B, const float* bias,
                     const float* scale, const float* res, float* C,
                     int M, int N, int K, int act) {
        cvt(A, ABF, M * K);
        gemm(ABF, B, bias, scale, res, C, M, N, K, K, act);
    };

    // -- one-time weight conversions (bf16 N x K) --
    cvt(c1_w, c1bf, 4 * 128 * 256);
    cvt(c3_w, c3bf, 4 * 256 * 128);
    for (int i = 0; i < 4; i++) {
        cvtT(gmsg_w + (size_t)i * 65536, gmsgbf + (size_t)i * 65536, 256, 256);
        cvtT(gupd_w + (size_t)i * 131072, gtopbf + (size_t)i * 65536, 256, 256);
        cvtT(gupd_w + (size_t)i * 131072 + 65536, gbotbf + (size_t)i * 65536, 256, 256);
        cvtT(tin_w + (size_t)i * 196608, tinbf + (size_t)i * 196608, 256, 768);
        cvtT(tout_w + (size_t)i * 65536, toutbf + (size_t)i * 65536, 256, 256);
        cvtT(tf1_w + (size_t)i * 524288, tf1bf + (size_t)i * 524288, 256, 2048);
        cvtT(tf2_w + (size_t)i * 524288, tf2bf + (size_t)i * 524288, 2048, 256);
    }
    for (int j = 0; j < 24; j++) {
        cvtT(bqkv_w + (size_t)j * 196608, qkvbf + (size_t)j * 196608, 256, 768);
        cvtT(bout_w + (size_t)j * 65536, outbf + (size_t)j * 65536, 256, 256);
    }
    for (int i = 0; i < 8; i++) {
        cvtT(bf1_w + (size_t)i * 262144, f1bf + (size_t)i * 262144, 256, 1024);
        cvtT(bf2_w + (size_t)i * 262144, f2wbf + (size_t)i * 262144, 1024, 256);
    }
    cvtT(pol_w, polbf, 256, 2187);

    // -- fp32 transposes for direct convs --
    transpose_kernel<<<(256 * 936 + 255) / 256, 256, 0, stream>>>(stem_w, WT, 256, 936);
    for (int i = 0; i < 4; i++)
        transpose_kernel<<<(128 * 36 + 255) / 256, 256, 0, stream>>>(
            c2_w + (size_t)i * 128 * 36, WT2 + (size_t)i * 36 * 128, 128, 36);

    // -- stem --
    stem_conv_kernel<<<R, 256, 0, stream>>>(x_seq, WT, stem_s, stem_b, F);

    // -- CNN blocks --
    for (int i = 0; i < 4; i++) {
        gemmA(F, c1bf + (size_t)i * 32768, bn1_b + i * 128, bn1_s + i * 128,
              nullptr, T1, R, 128, 256, 1);
        gconv_kernel<<<R, 128, 0, stream>>>(T1, WT2 + (size_t)i * 36 * 128,
                                            bn2_s + i * 128, bn2_b + i * 128, T2);
        gemmA(T2, c3bf + (size_t)i * 32768, bn3_b + i * 256, bn3_s + i * 256,
              F, XN, R, 256, 128, 1);
        float* tmp = F; F = XN; XN = tmp;
    }

    posadd_kernel<<<R, 256, 0, stream>>>(F, pos);

    // -- GNN blocks --
    for (int i = 0; i < 4; i++) {
        mean81_kernel<<<NBOARD, 256, 0, stream>>>(F, MSG1);
        gemmA(MSG1, gmsgbf + (size_t)i * 65536, gmsg_b + i * 256, nullptr,
              nullptr, MSG2, NBOARD, 256, 256, 0);
        gemmA(F, gtopbf + (size_t)i * 65536, nullptr, nullptr,
              nullptr, XN, R, 256, 256, 0);
        gemmA(MSG2, gbotbf + (size_t)i * 65536, gupd_b + i * 256, nullptr,
              nullptr, MSG1, NBOARD, 256, 256, 0);
        gnn_fuse_kernel<<<R, 256, 0, stream>>>(F, XN, MSG1, gln_s + i * 256, gln_b + i * 256);
    }

    // -- board transformer blocks --
    for (int i = 0; i < 8; i++) {
        for (int a = 0; a < 3; a++) {
            int la = i * 4 + a;
            ln_kernel<<<R, 256, 0, stream>>>(F, bln_s + la * 256, bln_b + la * 256, XN);
            gemmA(XN, qkvbf + (size_t)(i * 3 + a) * 196608, bqkv_b + (i * 3 + a) * 768,
                  nullptr, nullptr, QKV, R, 768, 256, 0);
            attention_kernel<<<NBOARD * 8, 128, 0, stream>>>(QKV, pmask, OB, 81, a);
            gemmA(OB, outbf + (size_t)(i * 3 + a) * 65536, bout_b + (i * 3 + a) * 256,
                  nullptr, F, XN, R, 256, 256, 0);
            float* tmp = F; F = XN; XN = tmp;
        }
        int la = i * 4 + 3;
        ln_kernel<<<R, 256, 0, stream>>>(F, bln_s + la * 256, bln_b + la * 256, XN);
        gemmA(XN, f1bf + (size_t)i * 262144, bf1_b + i * 1024, nullptr,
              nullptr, T1, R, 1024, 256, 1);
        gemmA(T1, f2wbf + (size_t)i * 262144, bf2_b + i * 256, nullptr,
              F, XN, R, 256, 1024, 0);
        float* tmp = F; F = XN; XN = tmp;
    }

    // -- temporal transformer (Nb=64, L=8) --
    mean81_kernel<<<NBOARD, 256, 0, stream>>>(F, Y);
    for (int i = 0; i < 4; i++) {
        gemmA(Y, tinbf + (size_t)i * 196608, tin_b + i * 768, nullptr,
              nullptr, YQ, NBOARD, 768, 256, 0);
        attention_kernel<<<64 * 8, 128, 0, stream>>>(YQ, nullptr, YA, 8, 0);
        gemmA(YA, toutbf + (size_t)i * 65536, tout_b + i * 256, nullptr,
              Y, MSG1, NBOARD, 256, 256, 0);
        ln_kernel<<<NBOARD, 256, 0, stream>>>(MSG1, tln1_s + i * 256, tln1_b + i * 256, Y);
        gemmA(Y, tf1bf + (size_t)i * 524288, tf1_b + i * 2048, nullptr,
              nullptr, YT, NBOARD, 2048, 256, 1);
        gemmA(YT, tf2bf + (size_t)i * 524288, tf2_b + i * 256, nullptr,
              Y, MSG1, NBOARD, 256, 2048, 0);
        ln_kernel<<<NBOARD, 256, 0, stream>>>(MSG1, tln2_s + i * 256, tln2_b + i * 256, Y);
    }

    // -- heads: last timestep rows (b*8+7) via lda=2048 on bf16 copy of Y --
    float* logits = (float*)d_out;
    cvt(Y, ABF, NBOARD * 256);
    gemm(ABF + 7 * 256, polbf, pol_b, nullptr, nullptr, logits, 64, 2187, 256, 2048, 0);
    logsoftmax_kernel<<<64, 256, 0, stream>>>(logits, 2187);
    value_kernel<<<64, 64, 0, stream>>>(Y, vw1, vb1, vw2, vb2, (float*)d_out + 64 * 2187);
    aux_kernel<<<64, 192, 0, stream>>>(Y, aux_w, aux_b, (float*)d_out + 64 * 2187 + 64);
}
